// HybridTTLayer_67216238182706
// MI455X (gfx1250) — matrix operand order repaired and checked by reading, not compiled, not run on hardware
//
#include <hip/hip_runtime.h>
#include <hip/hip_bf16.h>
#include <math.h>
#include <stdint.h>

typedef __bf16 bf16;
typedef __bf16 v16bf __attribute__((ext_vector_type(16)));
typedef float  v8f   __attribute__((ext_vector_type(8)));
typedef unsigned int u32x4 __attribute__((ext_vector_type(4)));
typedef int          i32x4 __attribute__((ext_vector_type(4)));
typedef int          i32x8 __attribute__((ext_vector_type(8)));

union AFrag { v16bf v; unsigned int u[8]; };

#if __has_builtin(__builtin_amdgcn_tensor_load_to_lds) && __has_builtin(__builtin_amdgcn_s_wait_tensorcnt)
#define USE_TDM 1
#else
#define USE_TDM 0
#endif

static __device__ __forceinline__ float gelu_exact(float x) {
    return 0.5f * x * (1.0f + erff(x * 0.7071067811865475f));
}
static __device__ __forceinline__ v8f vzero8() {
    v8f z = {0.f, 0.f, 0.f, 0.f, 0.f, 0.f, 0.f, 0.f};
    return z;
}

// ---------------------------------------------------------------- weight cvt
__global__ void cvt_f32_bf16(const float* __restrict__ src, bf16* __restrict__ dst, int n) {
    int i = blockIdx.x * blockDim.x + threadIdx.x;
    int stride = gridDim.x * blockDim.x;
    for (; i < n; i += stride) dst[i] = (bf16)src[i];
}

// core1 [r=16][b=8][y=16][s=16] -> w1t[ys=y*16+s][rb=b*16+r] (bf16, row-major [256][128])
__global__ void pack_core1(const float* __restrict__ core1, bf16* __restrict__ w1t) {
    int t = blockIdx.x * blockDim.x + threadIdx.x;
    if (t >= 256 * 128) return;
    int rb = t & 127, ys = t >> 7;
    int b = rb >> 4, r = rb & 15;
    int y = ys >> 4, s = ys & 15;
    w1t[t] = (bf16)core1[((r * 8 + b) * 16 + y) * 16 + s];
}

// ---------------------------------------------------------------- layernorm
__global__ __launch_bounds__(256) void layernorm_bf16(
    const float* __restrict__ X, const float* __restrict__ gamma, const float* __restrict__ beta,
    bf16* __restrict__ lnOut, bf16* __restrict__ resOut, int D)
{
    __shared__ float red[256];
    const int row = blockIdx.x;
    const float* xr = X + (size_t)row * D;
    float s = 0.f, s2 = 0.f;
    for (int i = threadIdx.x; i < D; i += 256) { float v = xr[i]; s += v; s2 += v * v; }
    red[threadIdx.x] = s; __syncthreads();
    for (int off = 128; off > 0; off >>= 1) {
        if (threadIdx.x < off) red[threadIdx.x] += red[threadIdx.x + off];
        __syncthreads();
    }
    float mean = red[0] / (float)D;
    __syncthreads();
    red[threadIdx.x] = s2; __syncthreads();
    for (int off = 128; off > 0; off >>= 1) {
        if (threadIdx.x < off) red[threadIdx.x] += red[threadIdx.x + off];
        __syncthreads();
    }
    float var = red[0] / (float)D - mean * mean;
    float rstd = rsqrtf(var + 1e-6f);
    bf16* lo = lnOut + (size_t)row * D;
    bf16* ro = resOut + (size_t)row * D;
    for (int i = threadIdx.x; i < D; i += 256) {
        float v = xr[i];
        lo[i] = (bf16)((v - mean) * rstd * gamma[i] + beta[i]);
        ro[i] = (bf16)v;
    }
}

// ---------------------------------------------------------------- bf16 WMMA GEMM
// C[M,N] f32 = A[M,K]bf16 @ B[K,N]bf16 (+ epilogue), 128x128 tile, BK=32,
// ping-pong LDS double buffering. A tile staged by the Tensor Data Mover
// (wave 0 issues tensor_load_to_lds; TDM pad feature reproduces the padded
// 40-element LDS row stride), B tile staged synchronously with transpose.
// FUSE==0: C = acc + bias[col]
// FUSE==1: g = sigmoid(acc + bias[col]); C = g*gelu(denseIn) + (1-g)*residual
template <int FUSE>
__global__ __launch_bounds__(256) void gemm_bf16(
    const bf16* __restrict__ A, const bf16* __restrict__ B,
    const float* __restrict__ bias, const float* __restrict__ denseIn,
    const float* __restrict__ residual, float* __restrict__ C,
    int M, int N, int K)
{
    __shared__ bf16 As[2][128][40];  // 80B row stride (20 banks) -> conflict-free frag reads
    __shared__ bf16 Bs[2][128][40];  // transposed: Bs[buf][n][k]

    const int tid = threadIdx.x;
    const int lane = tid & 31;
    const int wave = tid >> 5;
    const int wy = wave >> 1;
    const int wx = wave & 1;
    const int m0 = blockIdx.y * 128;
    const int n0 = blockIdx.x * 128;
    const int half = lane >> 4;
    const int lm = lane & 15;
    const int ktiles = K >> 5;

    v8f acc[2][4];
    #pragma unroll
    for (int i = 0; i < 2; ++i)
        #pragma unroll
        for (int j = 0; j < 4; ++j) acc[i][j] = vzero8();

    auto stageB = [&](int buf, int kt) {
        const int k0 = kt << 5;
        for (int c = tid; c < 512; c += 256) {
            int k = c >> 4, seg = c & 15;
            uint4 q = *(const uint4*)(B + (size_t)(k0 + k) * N + n0 + seg * 8);
            const bf16* e = (const bf16*)&q;
            #pragma unroll
            for (int i = 0; i < 8; ++i) Bs[buf][seg * 8 + i][k] = e[i];
        }
        if (kt + 1 < ktiles)
            __builtin_prefetch(B + (size_t)((kt << 5) + 32 + (tid >> 4)) * N + n0 + (tid & 15) * 8, 0, 1);
    };

#if USE_TDM
    auto issueA = [&](int buf, int kt) {
        if (wave == 0) {
            unsigned lds = (unsigned)(uintptr_t)&As[buf][0][0];
            unsigned long long ga =
                (unsigned long long)(uintptr_t)(A + (size_t)m0 * K + ((size_t)kt << 5));
            u32x4 g0;
            g0[0] = 1u;
            g0[1] = lds;
            g0[2] = (unsigned)ga;
            g0[3] = ((unsigned)(ga >> 32) & 0x01FFFFFFu) | (2u << 30);
            i32x8 g1;
            g1[0] = (1 << 16) | (1 << 20) | (3 << 22) | (3 << 25);
            g1[1] = (int)(((unsigned)K & 0xFFFFu) << 16);
            g1[2] = (int)((((unsigned)K >> 16) & 0xFFFFu) |
                          (((unsigned)M & 0xFFFFu) << 16));
            g1[3] = (int)((((unsigned)M >> 16) & 0xFFFFu) | (32u << 16));
            g1[4] = 128;
            g1[5] = K;
            g1[6] = 0;
            g1[7] = 0;
            i32x4 g2 = {0, 0, 0, 0};
            i32x4 g3 = {0, 0, 0, 0};
#if defined(__clang_major__) && (__clang_major__ >= 23)
            i32x8 g4 = {0, 0, 0, 0, 0, 0, 0, 0};
            __builtin_amdgcn_tensor_load_to_lds(g0, g1, g2, g3, g4, 0);
#else
            __builtin_amdgcn_tensor_load_to_lds(g0, g1, g2, g3, 0);
#endif
        }
    };
    auto waitA = [&]() { if (wave == 0) __builtin_amdgcn_s_wait_tensorcnt(0); };
#else
    auto issueA = [&](int buf, int kt) {
        const int k0 = kt << 5;
        for (int c = tid; c < 512; c += 256) {
            int row = c >> 2, seg = c & 3;
            uint4 q = *(const uint4*)(A + (size_t)(m0 + row) * K + k0 + seg * 8);
            *(uint4*)&As[buf][row][seg * 8] = q;
        }
    };
    auto waitA = [&]() {};
#endif

    issueA(0, 0);
    stageB(0, 0);
    waitA();
    __syncthreads();

    for (int kt = 0; kt < ktiles; ++kt) {
        const int cur = kt & 1;
        if (kt + 1 < ktiles) {
            issueA(cur ^ 1, kt + 1);
            stageB(cur ^ 1, kt + 1);
        }

        AFrag af[2], bq[4];
        #pragma unroll
        for (int mt = 0; mt < 2; ++mt) {
            int m = wy * 32 + mt * 16 + lm;
            #pragma unroll
            for (int v = 0; v < 8; ++v) {
                int k = (half << 4) + (v << 1);
                af[mt].u[v] = *(const unsigned int*)&As[cur][m][k];
            }
        }
        #pragma unroll
        for (int nt = 0; nt < 4; ++nt) {
            int n = wx * 64 + nt * 16 + lm;
            #pragma unroll
            for (int v = 0; v < 8; ++v) {
                int k = (half << 4) + (v << 1);
                bq[nt].u[v] = *(const unsigned int*)&Bs[cur][n][k];
            }
        }
        #pragma unroll
        for (int mt = 0; mt < 2; ++mt)
            #pragma unroll
            for (int nt = 0; nt < 4; ++nt)
                acc[mt][nt] = __builtin_amdgcn_wmma_f32_16x16x32_bf16(
                    false, af[mt].v, false, bq[nt].v, (short)0, acc[mt][nt], false, false);

        waitA();
        __syncthreads();
    }

    #pragma unroll
    for (int mt = 0; mt < 2; ++mt) {
        #pragma unroll
        for (int nt = 0; nt < 4; ++nt) {
            int col = n0 + wx * 64 + nt * 16 + lm;
            #pragma unroll
            for (int j = 0; j < 8; ++j) {
                int row = m0 + wy * 32 + mt * 16 + half * 8 + j;
                size_t idx = (size_t)row * N + col;
                float a = acc[mt][nt][j];
                if (FUSE == 0) {
                    C[idx] = a + bias[col];
                } else {
                    float g = 1.f / (1.f + expf(-(a + bias[col])));
                    C[idx] = g * gelu_exact(denseIn[idx]) + (1.f - g) * residual[idx];
                }
            }
        }
    }
}

// ---------------------------------------------------------------- fused TT chain
// Per n:  for x in 0..7:
//   T1 (VALU): a2[c][b*16+r] = sum_a h[a,b,c]*core0[a,x,r]          (16x128 bf16)
//   T2 (WMMA): o2 = a2 @ w1' (16x256, K=128); B-frags hoisted in VGPRs;
//              accumulators staged as bf16 into A3[(x,y)=x*16+y][(c,s)=c*16+s]
// then one batched T3 GEMM (WMMA): out[128,(z=32)] = A3[128,256] @ C2[256,32]
__global__ __launch_bounds__(256) void tt_fused(
    const float* __restrict__ hmid,   // [4096,1024]
    const float* __restrict__ core0,  // [8a][8x][16r]
    const bf16*  __restrict__ w1t,    // [256 ys][128 rb]
    const float* __restrict__ core2,  // [16s][16c][32z]
    float* __restrict__ htt)          // [4096,4096]
{
    __shared__ bf16  w1s[256][130];   // 65.0 KB  T2 B-matrix
    __shared__ bf16  c2sB[32][258];   // 16.1 KB  T3 B-matrix, transposed [z][cs]
    __shared__ float c0s[1024];       //  4.0 KB
    __shared__ float hs[1024];        //  4.0 KB
    __shared__ bf16  a2s[16][130];    //  4.1 KB  T2 A tile
    __shared__ bf16  a3s[128][258];   // 64.5 KB  T3 A tile

    const int tid = threadIdx.x;
    for (int i = tid; i < 256 * 128; i += 256) w1s[i >> 7][i & 127] = w1t[i];
    for (int i = tid; i < 32 * 256; i += 256) {
        int z = i >> 8, cs = i & 255, c = cs >> 4, s = cs & 15;
        c2sB[z][cs] = (bf16)core2[(s * 16 + c) * 32 + z];
    }
    for (int i = tid; i < 1024; i += 256) c0s[i] = core0[i];
    __syncthreads();

    const int lane = tid & 31;
    const int wave = tid >> 5;
    const int half = lane >> 4;
    const int lm = lane & 15;

    // hoist loop-invariant T2 B fragments: [col-tile][k-step], 64 VGPRs
    AFrag w1f[2][4];
    #pragma unroll
    for (int ct = 0; ct < 2; ++ct) {
        int coln = wave * 32 + ct * 16 + lm;
        #pragma unroll
        for (int kk = 0; kk < 4; ++kk)
            #pragma unroll
            for (int v = 0; v < 8; ++v)
                w1f[ct][kk].u[v] =
                    *(const unsigned int*)&w1s[coln][kk * 32 + (half << 4) + (v << 1)];
    }

    for (int n = blockIdx.x; n < 4096; n += gridDim.x) {
        for (int i = tid; i < 1024; i += 256) hs[i] = hmid[(size_t)n * 1024 + i];
        __syncthreads();

        for (int x = 0; x < 8; ++x) {
            // ---- T1
            #pragma unroll
            for (int e = 0; e < 8; ++e) {
                int o = e * 256 + tid;
                int c = o >> 7, rb = o & 127, b = rb >> 4, r = rb & 15;
                float sum = 0.f;
                #pragma unroll
                for (int a = 0; a < 8; ++a)
                    sum += hs[a * 128 + b * 16 + c] * c0s[(a * 8 + x) * 16 + r];
                a2s[c][rb] = (bf16)sum;
            }
            __syncthreads();
            // ---- T2 (A-frag only from LDS; B in registers)
            v8f acc0 = vzero8(), acc1 = vzero8();
            #pragma unroll
            for (int kk = 0; kk < 4; ++kk) {
                AFrag af;
                #pragma unroll
                for (int v = 0; v < 8; ++v) {
                    int k = kk * 32 + (half << 4) + (v << 1);
                    af.u[v] = *(const unsigned int*)&a2s[lm][k];
                }
                acc0 = __builtin_amdgcn_wmma_f32_16x16x32_bf16(
                    false, af.v, false, w1f[0][kk].v, (short)0, acc0, false, false);
                acc1 = __builtin_amdgcn_wmma_f32_16x16x32_bf16(
                    false, af.v, false, w1f[1][kk].v, (short)0, acc1, false, false);
            }
            // ---- stage T2 result transposed into T3's A tile:
            // acc tile ct covers y = wave*2+ct, lane lm = s, vgpr row c = half*8+j
            {
                int r0 = x * 16 + wave * 2;
                #pragma unroll
                for (int j = 0; j < 8; ++j) {
                    int cs = (half * 8 + j) * 16 + lm;
                    a3s[r0][cs]     = (bf16)acc0[j];
                    a3s[r0 + 1][cs] = (bf16)acc1[j];
                }
            }
            __syncthreads();
        }

        // ---- T3: batched GEMM [128 x 256] @ [256 x 32] on WMMA
        v8f oc0 = vzero8(), oc1 = vzero8();
        const int arow = wave * 16 + lm;
        #pragma unroll
        for (int kk = 0; kk < 8; ++kk) {
            AFrag af, b0, b1;
            #pragma unroll
            for (int v = 0; v < 8; ++v) {
                int k = kk * 32 + (half << 4) + (v << 1);
                af.u[v] = *(const unsigned int*)&a3s[arow][k];
            }
            #pragma unroll
            for (int v = 0; v < 8; ++v) {
                int k = kk * 32 + (half << 4) + (v << 1);
                b0.u[v] = *(const unsigned int*)&c2sB[lm][k];
                b1.u[v] = *(const unsigned int*)&c2sB[16 + lm][k];
            }
            oc0 = __builtin_amdgcn_wmma_f32_16x16x32_bf16(
                false, af.v, false, b0.v, (short)0, oc0, false, false);
            oc1 = __builtin_amdgcn_wmma_f32_16x16x32_bf16(
                false, af.v, false, b1.v, (short)0, oc1, false, false);
        }
        // store: out row = (x,y) = wave*16 + half*8 + j, cols z and z+16
        float* orow = htt + (size_t)n * 4096;
        #pragma unroll
        for (int j = 0; j < 8; ++j) {
            int ro = wave * 16 + half * 8 + j;
            orow[ro * 32 + lm]      = oc0[j];
            orow[ro * 32 + 16 + lm] = oc1[j];
        }
        __syncthreads();
    }
}

// ---------------------------------------------------------------- launch
extern "C" void kernel_launch(void* const* d_in, const int* in_sizes, int n_in,
                              void* d_out, int out_size, void* d_ws, size_t ws_size,
                              hipStream_t stream)
{
    (void)in_sizes; (void)n_in; (void)out_size; (void)ws_size;
    const float* x      = (const float*)d_in[0];
    const float* preG   = (const float*)d_in[1];
    const float* preB   = (const float*)d_in[2];
    const float* preDW  = (const float*)d_in[3];
    const float* preDB  = (const float*)d_in[4];
    const float* preGW  = (const float*)d_in[5];
    const float* preGB  = (const float*)d_in[6];
    const float* core0  = (const float*)d_in[7];
    const float* core1  = (const float*)d_in[8];
    const float* core2  = (const float*)d_in[9];
    const float* postG  = (const float*)d_in[10];
    const float* postB  = (const float*)d_in[11];
    const float* postDW = (const float*)d_in[12];
    const float* postDB = (const float*)d_in[13];
    const float* postGW = (const float*)d_in[14];
    const float* postGB = (const float*)d_in[15];
    float* out = (float*)d_out;

    char* ws = (char*)d_ws;
    const size_t MB = 1024ull * 1024ull;
    bf16* wPD  = (bf16*)(ws + 0);
    bf16* wPG  = (bf16*)(ws + 2 * MB);
    bf16* wQD  = (bf16*)(ws + 4 * MB);
    bf16* wQG  = (bf16*)(ws + 36 * MB);
    bf16* w1t  = (bf16*)(ws + 68 * MB);
    bf16* lnB  = (bf16*)(ws + 69 * MB);
    bf16* resB = (bf16*)(ws + 101 * MB);
    float* dO  = (float*)(ws + 133 * MB);
    float* hMid= (float*)(ws + 197 * MB);
    float* hTT = (float*)(ws + 213 * MB);

    cvt_f32_bf16<<<2048, 256, 0, stream>>>(preDW, wPD, 1024 * 1024);
    cvt_f32_bf16<<<2048, 256, 0, stream>>>(preGW, wPG, 1024 * 1024);
    cvt_f32_bf16<<<4096, 256, 0, stream>>>(postDW, wQD, 4096 * 4096);
    cvt_f32_bf16<<<4096, 256, 0, stream>>>(postGW, wQG, 4096 * 4096);
    pack_core1<<<128, 256, 0, stream>>>(core1, w1t);

    layernorm_bf16<<<4096, 256, 0, stream>>>(x, preG, preB, lnB, resB, 1024);
    gemm_bf16<0><<<dim3(8, 32), 256, 0, stream>>>(lnB, wPD, preDB, nullptr, nullptr, dO, 4096, 1024, 1024);
    gemm_bf16<1><<<dim3(8, 32), 256, 0, stream>>>(resB, wPG, preGB, dO, x, hMid, 4096, 1024, 1024);

    tt_fused<<<1024, 256, 0, stream>>>(hMid, core0, w1t, core2, hTT);

    layernorm_bf16<<<4096, 256, 0, stream>>>(hTT, postG, postB, lnB, resB, 4096);
    gemm_bf16<0><<<dim3(32, 32), 256, 0, stream>>>(lnB, wQD, postDB, nullptr, nullptr, dO, 4096, 4096, 4096);
    gemm_bf16<1><<<dim3(32, 32), 256, 0, stream>>>(resB, wQG, postGB, dO, hTT, out, 4096, 4096, 4096);
}
